// EdgeConvBlock_10282151707327
// MI455X (gfx1250) — matrix operand order repaired and checked by reading, not compiled, not run on hardware
//
#include <hip/hip_runtime.h>
#include <hip/hip_bf16.h>
#include <math.h>

#define CIN   64
#define COUT  64
#define TK    128   // 2*CIN, GEMM K dimension

typedef __attribute__((ext_vector_type(8)))  _Float16 v8h;
typedef __attribute__((ext_vector_type(16))) _Float16 v16h;
typedef __attribute__((ext_vector_type(8)))  float    v8f;

// ---- float <-> monotonic-uint encoding for atomicMax on floats ----
__device__ __forceinline__ unsigned encF(float f) {
    unsigned b = __float_as_uint(f);
    return (b & 0x80000000u) ? ~b : (b | 0x80000000u);
}
__device__ __forceinline__ float decF(unsigned u) {
    return __uint_as_float((u & 0x80000000u) ? (u & 0x7FFFFFFFu) : ~u);
}
__device__ __forceinline__ bool finiteF(float v) {
    return (__float_as_uint(v) & 0x7F800000u) != 0x7F800000u;
}

// ============================================================
// Pass 0: init agg to encoded(-inf); zero reduction scratch.
// ============================================================
__global__ __launch_bounds__(256) void init_kernel(unsigned* __restrict__ agg,
                                                   double* __restrict__ red,
                                                   int total) {
    int i = blockIdx.x * 256 + threadIdx.x;
    if (i < total) agg[i] = 0x007FFFFFu;   // encF(-inf)
    if (i < 2)     red[i] = 0.0;
}

// ============================================================
// Pass 1: global sum / sumsq of (x[src]-x[dst]) over E*64 elems
// ============================================================
__global__ __launch_bounds__(256) void reduce_std_kernel(
        const float* __restrict__ x, const long long* __restrict__ ei,
        double* __restrict__ red, int E) {
    __shared__ double ss[256];
    __shared__ double sq[256];
    int tid = threadIdx.x;
    int e = blockIdx.x * 256 + tid;
    float ls = 0.f, lq = 0.f;
    if (e < E) {
        long long s = ei[e];
        long long d = ei[E + e];
        const float4* pj = (const float4*)(x + s * CIN);
        const float4* pi = (const float4*)(x + d * CIN);
#pragma unroll
        for (int k = 0; k < CIN / 4; ++k) {
            float4 a = pj[k], b = pi[k];
            float d0 = a.x - b.x, d1 = a.y - b.y, d2 = a.z - b.z, d3 = a.w - b.w;
            ls += (d0 + d1) + (d2 + d3);
            lq += (d0 * d0 + d1 * d1) + (d2 * d2 + d3 * d3);
        }
    }
    ss[tid] = (double)ls;
    sq[tid] = (double)lq;
    __syncthreads();
    for (int o = 128; o > 0; o >>= 1) {
        if (tid < o) { ss[tid] += ss[tid + o]; sq[tid] += sq[tid + o]; }
        __syncthreads();
    }
    if (tid == 0) {
        atomicAdd(&red[0], ss[0]);
        atomicAdd(&red[1], sq[0]);
    }
}

// ============================================================
// Pass 2: fused gather -> affine -> concat -> GEMM (WMMA f16)
//         -> bias -> scatter atomic-max. 1 wave = 16 edges.
// ============================================================
__global__ __launch_bounds__(256) void edgeconv_wmma_kernel(
        const float* __restrict__ x, const long long* __restrict__ ei,
        const float* __restrict__ aw, const float* __restrict__ ab,
        const float* __restrict__ W, const float* __restrict__ bias,
        const double* __restrict__ red, unsigned* __restrict__ agg, int E) {
    // B matrix (weights) as f16, row-major [n][k]  (16 KB)
    __shared__ __align__(16) _Float16 Wlds[COUT * TK];
    // Per-wave A tiles: 16 edges x 128 features, f16  (32 KB)
    __shared__ __align__(16) _Float16 Alds[8][16 * TK];

    const int tid  = threadIdx.x;
    const int lane = tid & 31;
    const int wid  = tid >> 5;

    // ---- stage weights into LDS as f16 ----
    for (int i = tid; i < COUT * TK; i += 256) Wlds[i] = (_Float16)W[i];

    // ---- inv_std from pass-1 reduction (unbiased, ddof=1) ----
    double sum = red[0], sumsq = red[1];
    double n   = (double)E * (double)CIN;
    double var = (sumsq - sum * sum / n) / (n - 1.0);
    float inv_std = 1.0f / ((float)sqrt(var) + 1e-5f);

    // each lane owns channels 2*lane, 2*lane+1
    const int c0 = 2 * lane;
    const float aw0 = aw[c0], aw1 = aw[c0 + 1];
    const float ab0 = ab[c0], ab1 = ab[c0 + 1];

    const long long* srcI = ei;
    const long long* dstI = ei + E;
    const int base = blockIdx.x * 128 + wid * 16;

    _Float16* Arow = Alds[wid];

    // ---- stage A tile: [x_i | x_rel] as f16 in LDS ----
    for (int e = 0; e < 16; ++e) {
        int edge = base + e;
        float xi0 = 0.f, xi1 = 0.f, r0 = 0.f, r1 = 0.f;
        if (edge < E) {
            long long sIdx = srcI[edge];
            long long dIdx = dstI[edge];
            float2 vj = *(const float2*)(x + sIdx * CIN + c0);
            float2 vi = *(const float2*)(x + dIdx * CIN + c0);
            xi0 = vi.x; xi1 = vi.y;
            r0 = aw0 * ((vj.x - vi.x) * inv_std) + ab0;
            r1 = aw1 * ((vj.y - vi.y) * inv_std) + ab1;
        }
        _Float16* row = Arow + e * TK;
        row[c0]           = (_Float16)xi0;
        row[c0 + 1]       = (_Float16)xi1;
        row[CIN + c0]     = (_Float16)r0;
        row[CIN + c0 + 1] = (_Float16)r1;
    }
    __syncthreads();

    // ---- load A fragments (ISA 16-bit 16x32 layout) ----
    const int M    = lane & 15;
    const int aoff = (lane < 16) ? 0 : 8;   // K sub-offset for A
    v16h afrag[4];
#pragma unroll
    for (int c = 0; c < 4; ++c) {
        const _Float16* p = Arow + M * TK + c * 32 + aoff;
        v8h lo = *(const v8h*)p;          // K = c*32+aoff+0..7
        v8h hi = *(const v8h*)(p + 16);   // K = c*32+aoff+16..23
        afrag[c] = __builtin_shufflevector(lo, hi, 0, 1, 2, 3, 4, 5, 6, 7,
                                                   8, 9, 10, 11, 12, 13, 14, 15);
    }

    // ---- GEMM: 4 N-tiles x 4 K-chunks of v_wmma_f32_16x16x32_f16 ----
    const int koff = (lane < 16) ? 0 : 8;   // K sub-offset for B
    v8f acc[4];
#pragma unroll
    for (int t = 0; t < 4; ++t) {
        v8f z = {0.f, 0.f, 0.f, 0.f, 0.f, 0.f, 0.f, 0.f};
        acc[t] = z;
    }
#pragma unroll
    for (int t = 0; t < 4; ++t) {
#pragma unroll
        for (int c = 0; c < 4; ++c) {
            const _Float16* bp = Wlds + (t * 16 + M) * TK + c * 32 + koff;
            v8h blo = *(const v8h*)bp;        // K = c*32+koff+0..7
            v8h bhi = *(const v8h*)(bp + 16); // K = c*32+koff+16..23
            v16h bfrag = __builtin_shufflevector(blo, bhi, 0, 1, 2, 3, 4, 5, 6, 7,
                                                           8, 9, 10, 11, 12, 13, 14, 15);
            acc[t] = __builtin_amdgcn_wmma_f32_16x16x32_f16(
                false, afrag[c], false, bfrag, (short)0, acc[t], false, false);
        }
    }

    // ---- bias + scatter-max (D layout: M = r + 8*(lane>=16), N = lane%16) ----
    const int mhi = (lane >> 4) << 3;
    long long nodes[8];
#pragma unroll
    for (int r = 0; r < 8; ++r) {
        int edge = base + r + mhi;
        nodes[r] = (edge < E) ? dstI[edge] : (long long)-1;
    }
#pragma unroll
    for (int t = 0; t < 4; ++t) {
        int ncol = t * 16 + M;
        float bv = bias[ncol];
#pragma unroll
        for (int r = 0; r < 8; ++r) {
            if (nodes[r] >= 0) {
                float v = acc[t][r] + bv;
                atomicMax(agg + nodes[r] * COUT + ncol, encF(v));
            }
        }
    }
}

// ============================================================
// Pass 3: decode, -inf -> 0, LayerNorm over 64 feats, PReLU.
//         One wave32 per node; in-place over d_out.
// ============================================================
__global__ __launch_bounds__(256) void ln_prelu_kernel(
        unsigned* __restrict__ agg, const float* __restrict__ lnw,
        const float* __restrict__ lnb, const float* __restrict__ pa, int Nn) {
    int lane = threadIdx.x & 31;
    int wid  = threadIdx.x >> 5;
    int node = blockIdx.x * 8 + wid;
    if (node >= Nn) return;

    unsigned* row = agg + (long long)node * COUT;
    float v0 = decF(row[lane]);
    float v1 = decF(row[lane + 32]);
    if (!finiteF(v0)) v0 = 0.f;
    if (!finiteF(v1)) v1 = 0.f;

    float s = v0 + v1;
    float q = v0 * v0 + v1 * v1;
#pragma unroll
    for (int o = 16; o > 0; o >>= 1) {
        s += __shfl_xor(s, o, 32);
        q += __shfl_xor(q, o, 32);
    }
    float mu  = s * (1.0f / 64.0f);
    float var = q * (1.0f / 64.0f) - mu * mu;
    float rs  = rsqrtf(var + 1e-5f);
    float a   = pa[0];

    float h0 = (v0 - mu) * rs * lnw[lane]      + lnb[lane];
    float h1 = (v1 - mu) * rs * lnw[lane + 32] + lnb[lane + 32];
    h0 = (h0 > 0.f) ? h0 : a * h0;
    h1 = (h1 > 0.f) ? h1 : a * h1;

    float* out = (float*)row;
    out[lane]      = h0;
    out[lane + 32] = h1;
}

// ============================================================
extern "C" void kernel_launch(void* const* d_in, const int* in_sizes, int n_in,
                              void* d_out, int out_size, void* d_ws, size_t ws_size,
                              hipStream_t stream) {
    const float*     x    = (const float*)d_in[0];
    const long long* ei   = (const long long*)d_in[1];   // int64 edge_index [2,E]
    const float*     aw   = (const float*)d_in[2];
    const float*     ab   = (const float*)d_in[3];
    const float*     W    = (const float*)d_in[4];       // [64,128] row-major
    const float*     bias = (const float*)d_in[5];
    const float*     lnw  = (const float*)d_in[6];
    const float*     lnb  = (const float*)d_in[7];
    const float*     pa   = (const float*)d_in[8];

    const int E  = in_sizes[1] / 2;
    const int Nn = in_sizes[0] / CIN;

    unsigned* agg = (unsigned*)d_out;    // encoded scatter-max accumulator
    double*   red = (double*)d_ws;       // [sum, sumsq]

    const int total = Nn * COUT;
    init_kernel<<<(total + 255) / 256, 256, 0, stream>>>(agg, red, total);
    reduce_std_kernel<<<(E + 255) / 256, 256, 0, stream>>>(x, ei, red, E);
    edgeconv_wmma_kernel<<<(E + 127) / 128, 256, 0, stream>>>(
        x, ei, aw, ab, W, bias, red, agg, E);
    ln_prelu_kernel<<<(Nn + 7) / 8, 256, 0, stream>>>(agg, lnw, lnb, pa, Nn);
}
